// EdgeClassifier_20237885899320
// MI455X (gfx1250) — matrix operand order repaired and checked by reading, not compiled, not run on hardware
//
#include <hip/hip_runtime.h>

typedef __attribute__((ext_vector_type(16))) __bf16 v16bf;
typedef __attribute__((ext_vector_type(8)))  float  v8f;

#define N_NODES 50000
#define N_EDGES 800000
#define IN_NODE 5
#define IN_EDGE 29
#define HID 128
#define OUT 2

#if defined(__has_builtin)
#if __has_builtin(__builtin_amdgcn_global_load_async_to_lds_b128) && \
    __has_builtin(__builtin_amdgcn_s_wait_asynccnt)
#define USE_ASYNC_LDS 1
#endif
#endif
#ifndef USE_ASYNC_LDS
#define USE_ASYNC_LDS 0
#endif

#if USE_ASYNC_LDS
// Builtin expects: (int4 __device__ *, int4 __shared__ *, imm offset, imm cpol)
typedef int async_i4 __attribute__((vector_size(16)));
typedef __attribute__((address_space(1))) async_i4* g_async_p;
typedef __attribute__((address_space(3))) async_i4* l_async_p;
#endif

// Copy 16 bytes global -> LDS. Async (ASYNCcnt-tracked) when available.
__device__ __forceinline__ void stage16(const void* g, void* l) {
#if USE_ASYNC_LDS
  __builtin_amdgcn_global_load_async_to_lds_b128((g_async_p)g, (l_async_p)l, 0, 0);
#else
  *(uint4*)l = *(const uint4*)g;
#endif
}

__device__ __forceinline__ void stage_wait() {
#if USE_ASYNC_LDS
  __builtin_amdgcn_s_wait_asynccnt(0);
#endif
}

// ---------------------------------------------------------------------------
// Workspace layout (bytes, 256-aligned):
//   deg     : [N]          f32   @ 0
//   agg1    : [N,5]        f32   @ 200192
//   h1      : [N,128]      f32   @ 1200384
//   agg2    : [N,128]      f32   @ 26800384
//   h_nodes : [N,128]      f32   @ 52400384
//   W2bf    : [4][8][32][16] bf16 @ 78000384   (ktile, jtile, lane, elem)
//   Webf    : [8][32][16]    bf16 @ 78033152   (jtile, lane, elem) K padded 29->32
// ---------------------------------------------------------------------------

__global__ void zero_kernel(float* __restrict__ p, int n) {
  int i = blockIdx.x * blockDim.x + threadIdx.x;
  if (i < n) p[i] = 0.0f;
}

// Pre-swizzle weights into the CDNA5 WMMA bf16 B-matrix lane layout:
// B tile 32(K) x 16(N): lane L holds column N = L%16, elements e=0..7 hold
// K = 8*(L/16) + e, elements e=8..15 hold K = 16 + 8*(L/16) + (e-8).
__global__ void prep_weights(const float* __restrict__ W2, const float* __restrict__ We,
                             __bf16* __restrict__ W2bf, __bf16* __restrict__ Webf) {
  int tid = blockIdx.x * blockDim.x + threadIdx.x;
  if (tid < 32 * 512) {                       // W2: 4 ktiles * 8 jtiles * 512
    int tile = tid >> 9, within = tid & 511;
    int lane = within >> 4, e = within & 15;
    int kt = tile >> 3, jt = tile & 7;
    int k = kt * 32 + ((e >> 3) << 4) + ((lane >> 4) << 3) + (e & 7);
    int j = jt * 16 + (lane & 15);
    W2bf[tid] = (__bf16)W2[k * HID + j];
  } else if (tid < 32 * 512 + 8 * 512) {      // We: 1 ktile * 8 jtiles * 512
    int t2 = tid - 32 * 512;
    int tile = t2 >> 9, within = t2 & 511;
    int lane = within >> 4, e = within & 15;
    int k = ((e >> 3) << 4) + ((lane >> 4) << 3) + (e & 7);
    int j = tile * 16 + (lane & 15);
    Webf[t2] = (k < IN_EDGE) ? (__bf16)We[k * HID + j] : (__bf16)0.0f;
  }
}

// Edge scatter for conv1: deg[dst]+=1, agg1[dst] += node_feats[src]
__global__ void scatter1(const float* __restrict__ nf, const int* __restrict__ src,
                         const int* __restrict__ dst, float* __restrict__ agg1,
                         float* __restrict__ deg) {
  int e = blockIdx.x * blockDim.x + threadIdx.x;
  if (e >= N_EDGES) return;
  int s = src[e], d = dst[e];
  atomicAdd(&deg[d], 1.0f);
#pragma unroll
  for (int k = 0; k < IN_NODE; ++k)
    atomicAdd(&agg1[d * IN_NODE + k], nf[s * IN_NODE + k]);
}

// conv1: h1 = relu(((agg1 + feat)/(deg+1)) @ W1 + b1); K=5 -> plain VALU.
__global__ void conv1_kernel(const float* __restrict__ nf, const float* __restrict__ agg1,
                             const float* __restrict__ deg, const float* __restrict__ W1,
                             const float* __restrict__ b1, float* __restrict__ h1) {
  int tid = blockIdx.x * blockDim.x + threadIdx.x;
  int v = tid >> 7, j = tid & 127;
  if (v >= N_NODES) return;
  float invd = 1.0f / (deg[v] + 1.0f);
  float acc = b1[j];
#pragma unroll
  for (int k = 0; k < IN_NODE; ++k)
    acc += (agg1[v * IN_NODE + k] + nf[v * IN_NODE + k]) * invd * W1[k * HID + j];
  h1[(size_t)v * HID + j] = acc > 0.0f ? acc : 0.0f;
}

// Edge scatter for conv2: agg2[dst][f] += h1[src][f]; thread per (edge, feat).
__global__ void scatter2(const float* __restrict__ h1, const int* __restrict__ src,
                         const int* __restrict__ dst, float* __restrict__ agg2) {
  long long tid = (long long)blockIdx.x * blockDim.x + threadIdx.x;
  if (tid >= (long long)N_EDGES * HID) return;
  int e = (int)(tid >> 7), f = (int)(tid & 127);
  atomicAdd(&agg2[(size_t)dst[e] * HID + f], h1[(size_t)src[e] * HID + f]);
}

// conv2 via bf16 WMMA: h_nodes = relu(((agg2+h1)/(deg+1)) @ W2 + b2).
// W2 panel (32KB) staged once per block into LDS via async global->LDS copy;
// WMMA B operands then come from ds_load instead of per-wave VMEM round trips.
__global__ __launch_bounds__(256)
void conv2_wmma(const float* __restrict__ agg2, const float* __restrict__ h1,
                const float* __restrict__ deg, const __bf16* __restrict__ W2bf,
                const float* __restrict__ b2, float* __restrict__ h_nodes) {
  __shared__ __bf16 shW2[32 * 512];           // 32 KB: 32 tiles of 32x16 bf16
  for (int i = threadIdx.x; i < 2048; i += 256)   // 2048 x 16B chunks
    stage16((const char*)W2bf + i * 16, (char*)shW2 + i * 16);
  stage_wait();
  __syncthreads();

  const int lane = threadIdx.x & 31;
  const int wave = threadIdx.x >> 5;
  const int tile = blockIdx.x * 8 + wave;     // wave-uniform guard keeps EXEC full
  if (tile >= N_NODES / 16) return;
  const int m = lane & 15, half = lane >> 4;
  const int v = tile * 16 + m;
  const float invd = 1.0f / (deg[v] + 1.0f);
  const float* a2row = agg2 + (size_t)v * HID;
  const float* h1row = h1 + (size_t)v * HID;

  // A lane layout (16-bit, 16x32): lane L row M=L%16, elems 0..7 K=8*(L/16)+i,
  // elems 8..15 K=16+8*(L/16)+i.
  v16bf A[4];
#pragma unroll
  for (int kt = 0; kt < 4; ++kt) {
    int k0 = kt * 32 + half * 8;
    int k1 = kt * 32 + 16 + half * 8;
#pragma unroll
    for (int i = 0; i < 8; ++i) A[kt][i]     = (__bf16)((a2row[k0 + i] + h1row[k0 + i]) * invd);
#pragma unroll
    for (int i = 0; i < 8; ++i) A[kt][8 + i] = (__bf16)((a2row[k1 + i] + h1row[k1 + i]) * invd);
  }

  for (int jt = 0; jt < 8; ++jt) {
    float bias = b2[jt * 16 + m];
    v8f c;
#pragma unroll
    for (int i = 0; i < 8; ++i) c[i] = bias;
#pragma unroll
    for (int kt = 0; kt < 4; ++kt) {
      v16bf B = *(const v16bf*)(shW2 + (size_t)(kt * 8 + jt) * 512 + lane * 16);
      c = __builtin_amdgcn_wmma_f32_16x16x32_bf16(false, A[kt], false, B, (short)0, c,
                                                  false, false);
    }
    int j = jt * 16 + m;
#pragma unroll
    for (int r = 0; r < 8; ++r) {             // C layout: lanes0-15 M=r, lanes16-31 M=8+r
      int row = tile * 16 + half * 8 + r;
      float x = c[r];
      h_nodes[(size_t)row * HID + j] = x > 0.0f ? x : 0.0f;
    }
  }
}

// Fused edge embedding (WMMA) + classifier. 128 threads = 4 waves = 64 edges
// per block. We panel (8KB) async-staged to LDS; the 64x128 h_edges tile lives
// in LDS (32KB) so it never round-trips HBM (~0.8GB saved). Classifier: one
// wave per edge, coalesced gathers of h_nodes[src/dst], __shfl_xor reduction.
__global__ __launch_bounds__(128)
void edge_embed_classify(const float* __restrict__ edge_feats,
                         const __bf16* __restrict__ Webf, const float* __restrict__ be,
                         const float* __restrict__ h_nodes,
                         const int* __restrict__ src, const int* __restrict__ dst,
                         const float* __restrict__ Wc, const float* __restrict__ bc,
                         float* __restrict__ out) {
  __shared__ float  sh[64 * HID];             // 32 KB h_edges tile
  __shared__ __bf16 shWe[8 * 512];            // 8 KB We panel
  for (int i = threadIdx.x; i < 512; i += 128)    // 512 x 16B chunks
    stage16((const char*)Webf + i * 16, (char*)shWe + i * 16);
  stage_wait();
  __syncthreads();

  const int lane = threadIdx.x & 31;
  const int wave = threadIdx.x >> 5;
  const int m = lane & 15, half = lane >> 4;
  const int tile = blockIdx.x * 4 + wave;     // 16-edge tile per wave (E/16 exact)
  const int e0 = tile * 16;

  {  // ---- phase 1: h_edges = relu(edge_feats @ We + be) for this tile ----
    const float* er = edge_feats + (size_t)(e0 + m) * IN_EDGE;
    v16bf A;
#pragma unroll
    for (int i = 0; i < 8; ++i) A[i] = (__bf16)er[half * 8 + i];          // K 0..15 < 29
#pragma unroll
    for (int i = 0; i < 8; ++i) {
      int k = 16 + half * 8 + i;                                          // K 16..31
      A[8 + i] = (k < IN_EDGE) ? (__bf16)er[k] : (__bf16)0.0f;
    }
    for (int jt = 0; jt < 8; ++jt) {
      float bias = be[jt * 16 + m];
      v8f c;
#pragma unroll
      for (int i = 0; i < 8; ++i) c[i] = bias;
      v16bf B = *(const v16bf*)(shWe + (size_t)jt * 512 + lane * 16);
      c = __builtin_amdgcn_wmma_f32_16x16x32_bf16(false, A, false, B, (short)0, c,
                                                  false, false);
      int j = jt * 16 + m;
#pragma unroll
      for (int r = 0; r < 8; ++r) {
        int el = wave * 16 + half * 8 + r;
        float x = c[r];
        sh[el * HID + j] = x > 0.0f ? x : 0.0f;
      }
    }
  }
  __syncthreads();

  // ---- phase 2: out = [h_src || h_dst || h_edge] @ Wc + bc ----
  float wcs0[4], wcs1[4], wcd0[4], wcd1[4], wce0[4], wce1[4];
#pragma unroll
  for (int q = 0; q < 4; ++q) {
    int k = lane + 32 * q;
    wcs0[q] = Wc[k * OUT + 0];             wcs1[q] = Wc[k * OUT + 1];
    wcd0[q] = Wc[(HID + k) * OUT + 0];     wcd1[q] = Wc[(HID + k) * OUT + 1];
    wce0[q] = Wc[(2 * HID + k) * OUT + 0]; wce1[q] = Wc[(2 * HID + k) * OUT + 1];
  }
  const float bc0 = bc[0], bc1 = bc[1];

  for (int t = 0; t < 16; ++t) {
    int eg = e0 + t;
    int s = src[eg], d = dst[eg];
    float a0 = 0.0f, a1 = 0.0f;
#pragma unroll
    for (int q = 0; q < 4; ++q) {
      int k = lane + 32 * q;
      float hs = h_nodes[(size_t)s * HID + k];
      float hd = h_nodes[(size_t)d * HID + k];
      float he = sh[(wave * 16 + t) * HID + k];
      a0 += hs * wcs0[q] + hd * wcd0[q] + he * wce0[q];
      a1 += hs * wcs1[q] + hd * wcd1[q] + he * wce1[q];
    }
#pragma unroll
    for (int off = 16; off; off >>= 1) {
      a0 += __shfl_xor(a0, off, 32);
      a1 += __shfl_xor(a1, off, 32);
    }
    if (lane == 0) {
      out[(size_t)eg * OUT + 0] = a0 + bc0;
      out[(size_t)eg * OUT + 1] = a1 + bc1;
    }
  }
}

extern "C" void kernel_launch(void* const* d_in, const int* in_sizes, int n_in,
                              void* d_out, int out_size, void* d_ws, size_t ws_size,
                              hipStream_t stream) {
  (void)in_sizes; (void)n_in; (void)out_size; (void)ws_size;
  const float* nf  = (const float*)d_in[0];
  const float* ef  = (const float*)d_in[1];
  const int*   src = (const int*)d_in[2];
  const int*   dst = (const int*)d_in[3];
  const float* W1  = (const float*)d_in[4];
  const float* b1  = (const float*)d_in[5];
  const float* W2  = (const float*)d_in[6];
  const float* b2  = (const float*)d_in[7];
  const float* We  = (const float*)d_in[8];
  const float* be  = (const float*)d_in[9];
  const float* Wc  = (const float*)d_in[10];
  const float* bc  = (const float*)d_in[11];
  float* out = (float*)d_out;

  char* ws = (char*)d_ws;
  float*  deg     = (float*)(ws + 0);
  float*  agg1    = (float*)(ws + 200192);
  float*  h1      = (float*)(ws + 1200384);
  float*  agg2    = (float*)(ws + 26800384);
  float*  h_nodes = (float*)(ws + 52400384);
  __bf16* W2bf    = (__bf16*)(ws + 78000384);
  __bf16* Webf    = (__bf16*)(ws + 78033152);

  // zero deg+agg1 (contiguous incl. pad) and agg2
  zero_kernel<<<(300096 + 255) / 256, 256, 0, stream>>>(deg, 300096);
  zero_kernel<<<(N_NODES * HID + 255) / 256, 256, 0, stream>>>(agg2, N_NODES * HID);

  prep_weights<<<(32 * 512 + 8 * 512 + 255) / 256, 256, 0, stream>>>(W2, We, W2bf, Webf);

  scatter1<<<N_EDGES / 256, 256, 0, stream>>>(nf, src, dst, agg1, deg);

  conv1_kernel<<<(N_NODES * HID) / 256, 256, 0, stream>>>(nf, agg1, deg, W1, b1, h1);

  scatter2<<<(N_EDGES * HID) / 256, 256, 0, stream>>>(h1, src, dst, agg2);

  conv2_wmma<<<(N_NODES / 16 + 7) / 8, 256, 0, stream>>>(agg2, h1, deg, W2bf, b2, h_nodes);

  edge_embed_classify<<<N_EDGES / 64, 128, 0, stream>>>(ef, Webf, be, h_nodes, src, dst,
                                                        Wc, bc, out);
}
